// SelfAttention_14233521618953
// MI455X (gfx1250) — matrix operand order repaired and checked by reading, not compiled, not run on hardware
//
#include <hip/hip_runtime.h>
#include <hip/hip_bf16.h>
#include <math.h>

// Problem constants (B=16, S=2048, D=1024)
#define BB 16
#define SS 2048
#define DD 1024

typedef __bf16 bf16;
typedef __attribute__((ext_vector_type(16))) __bf16 v16bf;
typedef __attribute__((ext_vector_type(8)))  __bf16 v8bf;
typedef __attribute__((ext_vector_type(4)))  __bf16 v4bf;
typedef __attribute__((ext_vector_type(8)))  float  v8f;

union Frag16 {
    v16bf v;
    v8bf  h[2];
};

#if __has_builtin(__builtin_amdgcn_s_wait_asynccnt)
#define WAIT_ASYNC(n) __builtin_amdgcn_s_wait_asynccnt(n)
#else
#define WAIT_ASYNC(n) asm volatile("s_wait_asynccnt %0" :: "i"(n) : "memory")
#endif

// ---------------------------------------------------------------------------
// f32 -> bf16 convert (vectorized: 4 elems/thread)
// ---------------------------------------------------------------------------
__global__ __launch_bounds__(256)
void k_cvt_f32_bf16(const float* __restrict__ in, bf16* __restrict__ out, int n4)
{
    int i = blockIdx.x * blockDim.x + threadIdx.x;
    if (i < n4) {
        float4 f = ((const float4*)in)[i];
        v4bf o = { (bf16)f.x, (bf16)f.y, (bf16)f.z, (bf16)f.w };
        ((v4bf*)out)[i] = o;
    }
}

// ---------------------------------------------------------------------------
// f32 -> bf16 convert with transpose: out[n*ld + k] = in[k*ld + n]  (ld x ld)
// ---------------------------------------------------------------------------
__global__ __launch_bounds__(256)
void k_cvtT_f32_bf16(const float* __restrict__ in, bf16* __restrict__ out, int ld)
{
    int x = blockIdx.x * 16 + threadIdx.x; // n
    int y = blockIdx.y * 16 + threadIdx.y; // k
    out[(size_t)x * ld + y] = (bf16)in[(size_t)y * ld + x];
}

// ---------------------------------------------------------------------------
// Tiled bf16 WMMA GEMM:  C[M,N] = scale * A[M,K] * Bt[N,K]^T
//   A  : bf16 row-major, pitch lda
//   Bt : bf16 row-major [N][K] (i.e. B transposed), pitch ldb
//   C  : bf16 (OUT_BF16) or f32, pitch ldc
// Block = 256 threads (8 waves), block tile 128x128, K step 32.
// Wave w (mw=w>>2, nw=w&3) computes a 64x32 tile = 4x2 WMMA fragments.
// Tiles staged global->LDS with GLOBAL_LOAD_ASYNC_TO_LDS_B128 (ASYNCcnt),
// double-buffered with a 2-deep prefetch pipeline.
// ---------------------------------------------------------------------------
template <bool OUT_BF16>
__global__ __launch_bounds__(256)
void k_gemm_bf16(const bf16* __restrict__ A, const bf16* __restrict__ Bt,
                 void* __restrict__ Cp,
                 int lda, int ldb, int ldc, int K,
                 long long strideA, long long strideB, long long strideC,
                 float scale)
{
    // [buf][a/b][row][k], pitch 40 elems = 80 B (16B aligned chunks)
    __shared__ bf16 smem[2][2][128][40];

    const int tid    = threadIdx.x;
    const int lane   = tid & 31;
    const int wave   = tid >> 5;          // 0..7
    const int mw     = wave >> 2;         // 0..1  (64-row half)
    const int nw     = wave & 3;          // 0..3  (32-col quarter)
    const int m_lane = lane & 15;
    const int hh     = lane >> 4;         // half-wave select

    const int n0 = blockIdx.x * 128;
    const int m0 = blockIdx.y * 128;

    A  += (size_t)blockIdx.z * strideA;
    Bt += (size_t)blockIdx.z * strideB;

    // staging assignment: thread t covers row t>>1, 16-elem half t&1 (32 B)
    const int srow  = tid >> 1;           // 0..127
    const int shalf = tid & 1;

    const bf16* ga = A  + (size_t)(m0 + srow) * lda + 16 * shalf;
    const bf16* gb = Bt + (size_t)(n0 + srow) * ldb + 16 * shalf;

    const unsigned aoff      = (unsigned)((srow * 40 + 16 * shalf) * 2);
    const unsigned boff      = aoff + 128u * 40u * 2u;
    const unsigned bufstride = 2u * 128u * 40u * 2u;   // bytes per buffer

    const int steps = K / 32;

    auto issue = [&](int step, int buf) {
        const bf16* pa = ga + (size_t)step * 32;
        const bf16* pb = gb + (size_t)step * 32;
        unsigned la = (unsigned)buf * bufstride + aoff;
        unsigned lb = (unsigned)buf * bufstride + boff;
        // inst offset is added to BOTH the global and the LDS address
        asm volatile("global_load_async_to_lds_b128 %0, %1, off"
                     :: "v"(la), "v"(pa) : "memory");
        asm volatile("global_load_async_to_lds_b128 %0, %1, off offset:16"
                     :: "v"(la), "v"(pa) : "memory");
        asm volatile("global_load_async_to_lds_b128 %0, %1, off"
                     :: "v"(lb), "v"(pb) : "memory");
        asm volatile("global_load_async_to_lds_b128 %0, %1, off offset:16"
                     :: "v"(lb), "v"(pb) : "memory");
    };

    issue(0, 0);
    issue(1, 1);   // steps >= 2 always (K multiple of 64 here)

    v8f acc[4][2] = {};

    for (int i = 0; i < steps; ++i) {
        // drain current tile's 4 async loads; keep next tile's in flight
        if (i < steps - 1) { WAIT_ASYNC(4); } else { WAIT_ASYNC(0); }
        __syncthreads();

        const int buf = i & 1;

        Frag16 af[4], bfr[2];
#pragma unroll
        for (int mi = 0; mi < 4; ++mi) {
            // A 16x32: lane(m,h): K in {8h..8h+7} and {16+8h..16+8h+7}
            const bf16* p = &smem[buf][0][64 * mw + 16 * mi + m_lane][8 * hh];
            af[mi].h[0] = *(const v8bf*)p;
            af[mi].h[1] = *(const v8bf*)(p + 16);
        }
#pragma unroll
        for (int ni = 0; ni < 2; ++ni) {
            // B 32x16: lane(n,h): K in {8h..8h+7} and {16+8h..16+8h+7}
            const bf16* p = &smem[buf][1][32 * nw + 16 * ni + m_lane][8 * hh];
            bfr[ni].h[0] = *(const v8bf*)p;
            bfr[ni].h[1] = *(const v8bf*)(p + 16);
        }

#pragma unroll
        for (int mi = 0; mi < 4; ++mi)
#pragma unroll
            for (int ni = 0; ni < 2; ++ni)
                acc[mi][ni] = __builtin_amdgcn_wmma_f32_16x16x32_bf16(
                    false, af[mi].v, false, bfr[ni].v,
                    (short)0, acc[mi][ni], false, false);

        __syncthreads();   // all waves done reading buf before it is refilled
        if (i + 2 < steps) issue(i + 2, buf);
    }

    // ---- epilogue: C/D f32 layout: VGPR r, lane l -> row r + 8*(l>>4), col l&15
#pragma unroll
    for (int mi = 0; mi < 4; ++mi)
#pragma unroll
        for (int ni = 0; ni < 2; ++ni) {
            const int col = n0 + 32 * nw + 16 * ni + m_lane;
#pragma unroll
            for (int r = 0; r < 8; ++r) {
                const int row = m0 + 64 * mw + 16 * mi + r + 8 * hh;
                const float val = acc[mi][ni][r] * scale;
                const size_t idx = (size_t)blockIdx.z * strideC + (size_t)row * ldc + col;
                if (OUT_BF16) ((bf16*)Cp)[idx]  = (bf16)val;
                else          ((float*)Cp)[idx] = val;
            }
        }
}

// ---------------------------------------------------------------------------
// Row softmax in place: one block per row of `ncols` floats
// ---------------------------------------------------------------------------
__global__ __launch_bounds__(256)
void k_softmax_rows(float* __restrict__ S, int ncols)
{
    __shared__ float red[256];
    float* row = S + (size_t)blockIdx.x * ncols;
    const int tid = threadIdx.x;

    float m = -INFINITY;
    for (int c = tid; c < ncols; c += 256) m = fmaxf(m, row[c]);
    red[tid] = m; __syncthreads();
    for (int s = 128; s > 0; s >>= 1) {
        if (tid < s) red[tid] = fmaxf(red[tid], red[tid + s]);
        __syncthreads();
    }
    m = red[0];
    __syncthreads();

    float sum = 0.f;
    for (int c = tid; c < ncols; c += 256) {
        float e = __expf(row[c] - m);
        row[c] = e;
        sum += e;
    }
    red[tid] = sum; __syncthreads();
    for (int s = 128; s > 0; s >>= 1) {
        if (tid < s) red[tid] += red[tid + s];
        __syncthreads();
    }
    const float inv = 1.0f / red[0];
    for (int c = tid; c < ncols; c += 256) row[c] *= inv;
}

// ---------------------------------------------------------------------------
// attn_weight[b,k] = sum_q P[b,q,k]   (coalesced column sums)
// ---------------------------------------------------------------------------
__global__ __launch_bounds__(256)
void k_colsum(const float* __restrict__ P, float* __restrict__ w)
{
    const int k = blockIdx.x * 256 + threadIdx.x;  // 0..SS-1
    const int b = blockIdx.y;
    const float* base = P + (size_t)b * SS * SS + k;
    float s = 0.f;
    for (int q = 0; q < SS; ++q) s += base[(size_t)q * SS];
    w[b * SS + k] = s;
}

// ---------------------------------------------------------------------------
// output[b,d] = sum_k attn_weight[b,k] * V[b,k,d]
// ---------------------------------------------------------------------------
__global__ __launch_bounds__(256)
void k_outv(const float* __restrict__ V, const float* __restrict__ w,
            float* __restrict__ out)
{
    const int d = blockIdx.x * 256 + threadIdx.x;  // 0..DD-1
    const int b = blockIdx.y;
    const float* vb = V + (size_t)b * SS * DD;
    const float* wb = w + b * SS;
    float s = 0.f;
    for (int k = 0; k < SS; ++k) s += wb[k] * vb[(size_t)k * DD + d];
    out[b * DD + d] = s;
}

// ---------------------------------------------------------------------------
extern "C" void kernel_launch(void* const* d_in, const int* in_sizes, int n_in,
                              void* d_out, int out_size, void* d_ws, size_t ws_size,
                              hipStream_t stream)
{
    (void)in_sizes; (void)n_in; (void)out_size; (void)ws_size;

    const float* X  = (const float*)d_in[0];   // [B,S,D]
    const float* Wq = (const float*)d_in[1];   // [D,D]
    const float* Wk = (const float*)d_in[2];
    const float* Wv = (const float*)d_in[3];

    char* ws = (char*)d_ws;
    size_t off = 0;
    bf16* Xbf = (bf16*)(ws + off);  off += (size_t)BB * SS * DD * 2;   // 64 MiB
    bf16* WqT = (bf16*)(ws + off);  off += (size_t)DD * DD * 2;        //  2 MiB
    bf16* WkT = (bf16*)(ws + off);  off += (size_t)DD * DD * 2;
    bf16* WvT = (bf16*)(ws + off);  off += (size_t)DD * DD * 2;
    bf16* Qbf = (bf16*)(ws + off);  off += (size_t)BB * SS * DD * 2;   // 64 MiB
    bf16* Kbf = (bf16*)(ws + off);  off += (size_t)BB * SS * DD * 2;   // 64 MiB
    float* Vf = (float*)(ws + off); off += (size_t)BB * SS * DD * 4;   // 128 MiB
    float* Sc = (float*)(ws + off); off += (size_t)BB * SS * SS * 4;   // 256 MiB

    float* out_main = (float*)d_out;                // [B,D]
    float* out_attw = (float*)d_out + BB * DD;      // [B,S]

    // 1) precision staging
    k_cvt_f32_bf16<<<(BB * SS * DD / 4) / 256, 256, 0, stream>>>(X, Xbf, BB * SS * DD / 4);
    {
        dim3 tb(16, 16), tg(DD / 16, DD / 16);
        k_cvtT_f32_bf16<<<tg, tb, 0, stream>>>(Wq, WqT, DD);
        k_cvtT_f32_bf16<<<tg, tb, 0, stream>>>(Wk, WkT, DD);
        k_cvtT_f32_bf16<<<tg, tb, 0, stream>>>(Wv, WvT, DD);
    }

    // 2) Q/K/V projections: [32768 x 1024] x [1024 x 1024]
    {
        dim3 g(DD / 128, (BB * SS) / 128, 1);
        k_gemm_bf16<true ><<<g, 256, 0, stream>>>(Xbf, WqT, Qbf, DD, DD, DD, DD, 0, 0, 0, 1.0f);
        k_gemm_bf16<true ><<<g, 256, 0, stream>>>(Xbf, WkT, Kbf, DD, DD, DD, DD, 0, 0, 0, 1.0f);
        k_gemm_bf16<false><<<g, 256, 0, stream>>>(Xbf, WvT, Vf,  DD, DD, DD, DD, 0, 0, 0, 1.0f);
    }

    // 3) scores = Q K^T / sqrt(D), batched over B (K stored row-major == B^T)
    {
        dim3 g(SS / 128, SS / 128, BB);
        k_gemm_bf16<false><<<g, 256, 0, stream>>>(
            Qbf, Kbf, Sc, DD, DD, SS, DD,
            (long long)SS * DD, (long long)SS * DD, (long long)SS * SS,
            0.03125f /* 1/sqrt(1024) */);
    }

    // 4) row softmax in place
    k_softmax_rows<<<BB * SS, 256, 0, stream>>>(Sc, SS);

    // 5) attn_weight = column sums of softmax  (second output)
    k_colsum<<<dim3(SS / 256, BB), 256, 0, stream>>>(Sc, out_attw);

    // 6) output[b,d] = attn_weight[b,:] . V[b,:,d]   (first output)
    k_outv<<<dim3(DD / 256, BB), 256, 0, stream>>>(Vf, out_attw, out_main);
}
